// ContextSelfSimilarity_37726992728563
// MI455X (gfx1250) — matrix operand order repaired and checked by reading, not compiled, not run on hardware
//
#include <hip/hip_runtime.h>
#include <hip/hip_bf16.h>
#include <stdint.h>

// Problem constants (from reference setup_inputs)
#define CIN  256
#define ACH  128
#define BB   8
#define HH   96
#define WW   96
#define HW   (HH*WW)      // 9216
#define NT   (BB*HW)      // 73728
#define KK   7
#define PADK 3

typedef __attribute__((ext_vector_type(16))) __bf16 v16bf;
typedef __attribute__((ext_vector_type(8)))  float  v8f;

union FragBF {
  v16bf bf;
  unsigned short s[16];
  uint4 q4[2];
};

// Native f32 -> bf16 (compiler emits hardware cvt, not a bit-twiddle chain)
__device__ __forceinline__ unsigned short f2bf(float f) {
  union { __bf16 h; unsigned short u; } t;
  t.h = (__bf16)f;
  return t.u;
}
__device__ __forceinline__ float bf2f(unsigned short h) {
  union { unsigned u; float f; } v; v.u = ((unsigned)h) << 16; return v.f;
}
__device__ __forceinline__ float bflo(unsigned u) {
  union { unsigned u; float f; } v; v.u = u << 16; return v.f;
}
__device__ __forceinline__ float bfhi(unsigned u) {
  union { unsigned u; float f; } v; v.u = u & 0xFFFF0000u; return v.f;
}

// ---------------------------------------------------------------------------
// Phase A0: pre-swizzle stacked weights [w_q; w_k] (256x256 f32) into bf16,
// laid out exactly as per-lane WMMA A-fragments:
//   wbuf[(((mtg*8 + ks)*32 + lane)*16 + j]
// where (16-bit A 16x32 ISA layout):
//   row   g = mtg*16 + (lane&15)
//   K     = ks*32 + j + (j&8) + (lane>=16 ? 8 : 0)
// Total 16*8*32*16 = 65536 bf16 = 128 KB.
// ---------------------------------------------------------------------------
__global__ void __launch_bounds__(256)
pack_weights(const float* __restrict__ wq,
             const float* __restrict__ wk,
             unsigned short* __restrict__ wbuf)
{
  const int flat = blockIdx.x * 256 + threadIdx.x;   // 0 .. 65535
  const int j    = flat & 15;
  const int lane = (flat >> 4) & 31;
  const int ks   = (flat >> 9) & 7;
  const int mtg  = (flat >> 12) & 15;

  const int g  = mtg * 16 + (lane & 15);
  const int K  = ks * 32 + j + (j & 8) + ((lane >= 16) ? 8 : 0);
  const float s = (g < ACH) ? wq[(size_t)g * CIN + K]
                            : wk[(size_t)(g - ACH) * CIN + K];
  wbuf[flat] = f2bf(s);
}

// ---------------------------------------------------------------------------
// Phase A: q/k projection as one bf16 WMMA GEMM.
//   D[m, n] = sum_c Wstk[m, c] * X[c, n],  m in [0,256), n = pixel, K = 256
// Output pixel-major, channel-contiguous bf16: ws[(b*HW+hw)*128 + ch]
// ---------------------------------------------------------------------------
#define BN 64              // pixels (N columns) per block
#define XSTRIDE 264        // LDS row stride (ushorts), 528B = 33*16B aligned

__global__ void __launch_bounds__(256)
qk_proj_wmma(const float* __restrict__ x,
             const unsigned short* __restrict__ wbuf,
             unsigned short* __restrict__ q_ws,
             unsigned short* __restrict__ k_ws)
{
  __shared__ unsigned short lds_x[BN][XSTRIDE];   // transposed: [pixel][channel]

  const int tid = threadIdx.x;
  const int n0  = blockIdx.x * BN;     // 9216 % 64 == 0 -> tile within one batch
  const int b   = n0 / HW;
  const int hw0 = n0 % HW;
  const float* xb = x + (size_t)b * CIN * HW + hw0;

  // Stage X tile: f32 -> bf16 (hw cvt), transpose to [pixel][channel]
  for (int i = tid; i < CIN * (BN / 4); i += 256) {
    const int c  = i / (BN / 4);
    const int cq = i % (BN / 4);
    const float4 v = *(const float4*)(xb + (size_t)c * HW + cq * 4);
    lds_x[cq * 4 + 0][c] = f2bf(v.x);
    lds_x[cq * 4 + 1][c] = f2bf(v.y);
    lds_x[cq * 4 + 2][c] = f2bf(v.z);
    lds_x[cq * 4 + 3][c] = f2bf(v.w);
  }
  __syncthreads();

  const int wave   = tid >> 5;
  const int lane   = tid & 31;
  const int l15    = lane & 15;
  const int hiHalf = (lane >= 16) ? 1 : 0;
  const int bbase  = hiHalf ? 8 : 0;    // B frag K base (16-bit B 32x16 layout)

  v8f acc[2][4];
  const v8f vzero = {0.f, 0.f, 0.f, 0.f, 0.f, 0.f, 0.f, 0.f};
#pragma unroll
  for (int mi = 0; mi < 2; ++mi)
#pragma unroll
    for (int ni = 0; ni < 4; ++ni) acc[mi][ni] = vzero;

  // Pre-swizzled A-fragment pointers: contiguous 32B per lane per K-step
  const unsigned short* wp[2];
#pragma unroll
  for (int mi = 0; mi < 2; ++mi)
    wp[mi] = wbuf + ((size_t)((wave * 2 + mi) * 8) * 32 + lane) * 16;

#pragma unroll 1
  for (int ks = 0; ks < CIN / 32; ++ks) {
    const int K0 = ks * 32;

    FragBF afr[2];
#pragma unroll
    for (int mi = 0; mi < 2; ++mi) {
      const unsigned short* p = wp[mi] + (size_t)ks * 32 * 16;
      afr[mi].q4[0] = *(const uint4*)(p);
      afr[mi].q4[1] = *(const uint4*)(p + 8);
    }

    FragBF bfr[4];
#pragma unroll
    for (int ni = 0; ni < 4; ++ni) {
      const unsigned short* p = &lds_x[ni * 16 + l15][K0 + bbase];
      bfr[ni].q4[0] = *(const uint4*)(p);
      bfr[ni].q4[1] = *(const uint4*)(p + 16);
    }

#pragma unroll
    for (int mi = 0; mi < 2; ++mi)
#pragma unroll
      for (int ni = 0; ni < 4; ++ni)
        acc[mi][ni] = __builtin_amdgcn_wmma_f32_16x16x32_bf16(
            false, afr[mi].bf, false, bfr[ni].bf,
            (short)0, acc[mi][ni], false, false);
  }

  // Store: D layout => lane holds 8 consecutive channels of one pixel -> b128
#pragma unroll
  for (int mi = 0; mi < 2; ++mi) {
    const int mtg = wave * 2 + mi;
    const int ch0 = mtg * 16 + (hiHalf ? 8 : 0);
    unsigned short* dst = (ch0 < ACH) ? q_ws : k_ws;
    const int chb = (ch0 < ACH) ? ch0 : (ch0 - ACH);
#pragma unroll
    for (int ni = 0; ni < 4; ++ni) {
      const int pix = hw0 + ni * 16 + l15;
      unsigned short hs[8];
#pragma unroll
      for (int v = 0; v < 8; ++v) hs[v] = f2bf(acc[mi][ni][v]);
      uint4 pk;
      pk.x = (unsigned)hs[0] | ((unsigned)hs[1] << 16);
      pk.y = (unsigned)hs[2] | ((unsigned)hs[3] << 16);
      pk.z = (unsigned)hs[4] | ((unsigned)hs[5] << 16);
      pk.w = (unsigned)hs[6] | ((unsigned)hs[7] << 16);
      *(uint4*)(dst + ((size_t)b * HW + pix) * ACH + chb) = pk;
    }
  }
}

// ---------------------------------------------------------------------------
// Phase B: windowed similarity + positional bias. One block per (b, h).
// ---------------------------------------------------------------------------
#define QSTRIDE 136        // ushorts; 272B row stride -> uint4-aligned rows
#define KCOLS   102        // W + 2*PADK

__global__ void __launch_bounds__(256)
sim_window(const unsigned short* __restrict__ q_ws,
           const unsigned short* __restrict__ k_ws,
           const float* __restrict__ rel_h,
           const float* __restrict__ rel_w,
           float* __restrict__ out)
{
  __shared__ unsigned short q_lds[WW][QSTRIDE];
  __shared__ unsigned short k_lds[KCOLS][QSTRIDE];
  __shared__ float bias_h[WW][KK];
  __shared__ float bias_w[WW][KK];

  const int tid = threadIdx.x;
  const int b   = blockIdx.x / HH;
  const int h   = blockIdx.x % HH;

  // Stage q row: 96 px * 128 ch bf16 as b128 chunks
  const unsigned short* qrow = q_ws + ((size_t)b * HW + h * WW) * ACH;
  for (int ch = tid; ch < WW * ACH / 8; ch += 256) {  // 1536 chunks
    const int pix = ch >> 4;
    const int c8  = (ch & 15) * 8;
    *(uint4*)&q_lds[pix][c8] = *(const uint4*)(qrow + pix * ACH + c8);
  }
  __syncthreads();

  // Positional biases: bias_h from first 64 channels, bias_w from last 64
  for (int o = tid; o < WW * KK; o += 256) {
    const int pix = o / KK, kk = o % KK;
    float ah = 0.f, aw = 0.f;
#pragma unroll 8
    for (int c = 0; c < 64; ++c) {
      ah = fmaf(bf2f(q_lds[pix][c]),      rel_h[c * KK + kk], ah);
      aw = fmaf(bf2f(q_lds[pix][64 + c]), rel_w[c * KK + kk], aw);
    }
    bias_h[pix][kk] = ah;
    bias_w[pix][kk] = aw;
  }

  const size_t outbase = ((size_t)(b * HH + h)) * WW * KK * KK;

  for (int ki = 0; ki < KK; ++ki) {
    __syncthreads();   // protect k_lds reuse (and first-iter bias visibility)
    const int hr = h + ki - PADK;
    // Stage padded k row (cols -3..98), zero outside the image
    for (int ch = tid; ch < KCOLS * ACH / 8; ch += 256) {  // 1632 chunks
      const int pl = ch >> 4;
      const int c8 = (ch & 15) * 8;
      const int wc = pl - PADK;
      uint4 val; val.x = 0u; val.y = 0u; val.z = 0u; val.w = 0u;
      if (hr >= 0 && hr < HH && wc >= 0 && wc < WW)
        val = *(const uint4*)(k_ws + ((size_t)b * HW + hr * WW + wc) * ACH + c8);
      *(uint4*)&k_lds[pl][c8] = val;
    }
    __syncthreads();

    for (int o = tid; o < WW * KK; o += 256) {
      const int w  = o / KK;
      const int kj = o % KK;
      const unsigned short* qp = &q_lds[w][0];
      const unsigned short* kp = &k_lds[w + kj][0];
      float acc = 0.f;
#pragma unroll
      for (int c = 0; c < ACH; c += 8) {
        const uint4 qv = *(const uint4*)(qp + c);
        const uint4 kv = *(const uint4*)(kp + c);
        acc = fmaf(bflo(qv.x), bflo(kv.x), acc);
        acc = fmaf(bfhi(qv.x), bfhi(kv.x), acc);
        acc = fmaf(bflo(qv.y), bflo(kv.y), acc);
        acc = fmaf(bfhi(qv.y), bfhi(kv.y), acc);
        acc = fmaf(bflo(qv.z), bflo(kv.z), acc);
        acc = fmaf(bfhi(qv.z), bfhi(kv.z), acc);
        acc = fmaf(bflo(qv.w), bflo(kv.w), acc);
        acc = fmaf(bfhi(qv.w), bfhi(kv.w), acc);
      }
      out[outbase + (size_t)w * KK * KK + ki * KK + kj] =
          acc + bias_h[w][ki] + bias_w[w][kj];
    }
  }
}

// ---------------------------------------------------------------------------
extern "C" void kernel_launch(void* const* d_in, const int* in_sizes, int n_in,
                              void* d_out, int out_size, void* d_ws, size_t ws_size,
                              hipStream_t stream) {
  (void)in_sizes; (void)n_in; (void)out_size; (void)ws_size;
  const float* x  = (const float*)d_in[0];
  const float* wq = (const float*)d_in[1];
  const float* wk = (const float*)d_in[2];
  const float* rh = (const float*)d_in[3];
  const float* rw = (const float*)d_in[4];
  float* out = (float*)d_out;

  unsigned short* q_ws = (unsigned short*)d_ws;            // NT*128 bf16
  unsigned short* k_ws = q_ws + (size_t)NT * ACH;          // NT*128 bf16
  unsigned short* wbuf = k_ws + (size_t)NT * ACH;          // 65536 bf16 (128 KB)

  pack_weights<<<256, 256, 0, stream>>>(wq, wk, wbuf);
  qk_proj_wmma<<<NT / BN, 256, 0, stream>>>(x, wbuf, q_ws, k_ws);
  sim_window<<<BB * HH, 256, 0, stream>>>(q_ws, k_ws, rh, rw, out);
}
